// pwc_model_24558622998865
// MI455X (gfx1250) — matrix operand order repaired and checked by reading, not compiled, not run on hardware
//
#include <hip/hip_runtime.h>

typedef __attribute__((ext_vector_type(16))) _Float16 v16h;
typedef __attribute__((ext_vector_type(8)))  float    v8f;

#define WAVES_PER_BLOCK 4

// Per-wave shared-memory working set (one center per wave).
struct WaveSmem {
  float d2[160];        // masked squared distances for window candidates
  int   selidx[32];     // flat feat index of selected neighbor (layer 1)
  float selval[32];     // validity mask of selected neighbor
  float bufA[32][32];   // 32 neighbors x 32 channels (zero padded), ping
  float bufB[32][32];   // pong
};

// One MLP layer (CIN -> COUT, CIN,COUT <= 32) applied to 32 rows via
// v_wmma_f32_16x16x32_f16: 2 M-tiles (rows 0-15, 16-31), ceil(COUT/16) N-tiles.
// Fragment layouts per CDNA5 ISA 7.12.2:
//   A 16x32 f16 : lane<16 row M=lane holds K {0..7,16..23}; lane>=16 row M=lane-16
//                 holds K {8..15,24..31}; element h -> K = (h<8?h:h-8+16)+(hi?8:0)
//   B 32x16 f16 : col N=lane&15; same K order as A: lanes 0-15 K {0..7,16..23}; lanes 16-31 K {8..15,24..31}
//   C/D 16x16 f32: VGPR v, lane<16 -> (M=v,  N=lane); lane>=16 -> (M=v+8, N=lane-16)
template<int CIN, int COUT>
__device__ inline void mlp_layer(const float* __restrict__ W,
                                 const float* __restrict__ Bias,
                                 const float (*gin)[32], float (*gout)[32],
                                 int lane)
{
  const bool hi = lane >= 16;
  const int  m  = lane & 15;
  const int  kbase = hi ? 8 : 0;

  v16h a0, a1;
  const float* r0 = gin[m];
  const float* r1 = gin[16 + m];
#pragma unroll
  for (int h = 0; h < 8; ++h) {
    a0[h]     = (_Float16)r0[kbase + h];
    a0[8 + h] = (_Float16)r0[16 + kbase + h];
    a1[h]     = (_Float16)r1[kbase + h];
    a1[8 + h] = (_Float16)r1[16 + kbase + h];
  }

  const int NT = (COUT + 15) / 16;
#pragma unroll
  for (int u = 0; u < NT; ++u) {
    const int n  = u * 16 + m;          // output channel this lane produces
    const int kb = hi ? 16 : 0;
    v16h bf;
#pragma unroll
    for (int h = 0; h < 16; ++h) {
      const int k = (h < 8 ? h : h + 8) + (hi ? 8 : 0); (void)kb;
      const float wv = (k < CIN && n < COUT) ? W[k * COUT + n] : 0.0f;
      bf[h] = (_Float16)wv;
    }
    const float bias = (n < COUT) ? Bias[n] : 0.0f;

    v8f cz = {};
    v8f d0 = __builtin_amdgcn_wmma_f32_16x16x32_f16(false, a0, false, bf,
                                                    (short)0, cz, false, false);
    v8f d1 = __builtin_amdgcn_wmma_f32_16x16x32_f16(false, a1, false, bf,
                                                    (short)0, cz, false, false);
#pragma unroll
    for (int v = 0; v < 8; ++v) {
      const int mrow = v + (hi ? 8 : 0);
      gout[mrow][n]      = (n < COUT) ? fmaxf(d0[v] + bias, 0.0f) : 0.0f;
      gout[16 + mrow][n] = (n < COUT) ? fmaxf(d1[v] + bias, 0.0f) : 0.0f;
    }
  }
  if (NT == 1) {
    // keep K-padding columns of the next layer's A-matrix zero
#pragma unroll
    for (int c = 16; c < 32; ++c) gout[lane][c] = 0.0f;
  }
}

// ---------------- Layer 0: 9x15 window on 32x1792 grid, K=32, MLP 6->8->8->16
__global__ void __launch_bounds__(128)
sa_layer0_kernel(const float* __restrict__ xyz,   // (B,64,1792,3); xyz1 rows = 2h
                 const float* __restrict__ w0, const float* __restrict__ b0,
                 const float* __restrict__ w1, const float* __restrict__ b1,
                 const float* __restrict__ w2, const float* __restrict__ b2,
                 float* __restrict__ ctr0,        // (B,8,224,3)
                 float* __restrict__ f0)          // (B,8,224,16)
{
  __shared__ WaveSmem sm[WAVES_PER_BLOCK];
  const int lane = threadIdx.x & 31;
  const int wv   = threadIdx.x >> 5;
  WaveSmem& S = sm[wv];

  const int cid = blockIdx.x * WAVES_PER_BLOCK + wv;      // 32*8*224 centers
  const int b  = cid / (8 * 224);
  const int r  = cid % (8 * 224);
  const int ic = r / 224;
  const int jc = r % 224;
  const int hcen = ic * 4;   // row in xyz1 (H=32)
  const int wcen = jc * 8;

  const long cbase = (((long)b * 64 + 2 * hcen) * 1792 + wcen) * 3;
  const float cx = xyz[cbase + 0], cy = xyz[cbase + 1], cz = xyz[cbase + 2];

  // ---- Phase 1: gather window, masked d^2 to LDS, keep rel in registers
  float dv[5], rlx[5], rly[5], rlz[5];
#pragma unroll
  for (int c = 0; c < 5; ++c) {
    const int p = c * 32 + lane;           // 0..159, real candidates < 135
    float d = 3.0e38f, rx = 0.f, ry = 0.f, rz = 0.f;
    if (p < 135) {
      const int kh = p / 15, kw = p % 15;
      int h = hcen + kh - 4; h = h < 0 ? 0 : (h > 31 ? 31 : h);
      int w = wcen + kw - 7; w = w < 0 ? 0 : (w > 1791 ? 1791 : w);
      const long gb = (((long)b * 64 + 2 * h) * 1792 + w) * 3;
      rx = xyz[gb + 0] - cx; ry = xyz[gb + 1] - cy; rz = xyz[gb + 2] - cz;
      const float d2 = rx * rx + ry * ry + rz * rz;
      d = (d2 < 0.75f) ? d2 : 1.0e10f;     // thresholded mask value (ref semantics)
    }
    S.d2[p] = d;
    dv[c] = d; rlx[c] = rx; rly[c] = ry; rlz[c] = rz;
  }
  __syncthreads();

  // ---- Phase 2: rank select (matches top_k tie-break: ascending d2, low index)
  int rnk[5] = {0, 0, 0, 0, 0};
  for (int q = 0; q < 135; ++q) {
    const float dq = S.d2[q];
#pragma unroll
    for (int c = 0; c < 5; ++c) {
      const int p = c * 32 + lane;
      rnk[c] += (dq < dv[c] || (dq == dv[c] && q < p)) ? 1 : 0;
    }
  }
#pragma unroll
  for (int c = 0; c < 5; ++c) {
    if (rnk[c] < 32) {
      const float vf = (dv[c] < 9.9e9f) ? 1.0f : 0.0f;
      float* row = S.bufA[rnk[c]];
      row[0] = rlx[c] * vf;        row[1] = rly[c] * vf;        row[2] = rlz[c] * vf;
      row[3] = (rlx[c] + cx) * vf; row[4] = (rly[c] + cy) * vf; row[5] = (rlz[c] + cz) * vf;
    }
  }
#pragma unroll
  for (int c = 6; c < 32; ++c) S.bufA[lane][c] = 0.0f;   // K padding
  __syncthreads();

  // ---- Phase 3: shared MLP on the 32x6 tile via WMMA
  mlp_layer<6, 8>(w0, b0, S.bufA, S.bufB, lane);  __syncthreads();
  mlp_layer<8, 8>(w1, b1, S.bufB, S.bufA, lane);  __syncthreads();
  mlp_layer<8, 16>(w2, b2, S.bufA, S.bufB, lane); __syncthreads();

  // ---- Phase 4: max-pool over the 32 neighbors (lane = channel)
  float mx = S.bufB[0][lane];
#pragma unroll
  for (int rr = 1; rr < 32; ++rr) mx = fmaxf(mx, S.bufB[rr][lane]);

  const long obase = (((long)b * 8 + ic) * 224 + jc);
  if (lane < 16) f0[obase * 16 + lane] = mx;
  if (lane < 3)  ctr0[obase * 3 + lane] = (lane == 0 ? cx : (lane == 1 ? cy : cz));
}

// ---------------- Layer 1: 7x11 window on 8x224 grid, K=32, MLP 19->16->16->32
__global__ void __launch_bounds__(128)
sa_layer1_kernel(const float* __restrict__ ctr0,  // (B,8,224,3)
                 const float* __restrict__ f0,    // (B,8,224,16)
                 const float* __restrict__ w0, const float* __restrict__ b0,
                 const float* __restrict__ w1, const float* __restrict__ b1,
                 const float* __restrict__ w2, const float* __restrict__ b2,
                 float* __restrict__ out)         // (B,896,32)
{
  __shared__ WaveSmem sm[WAVES_PER_BLOCK];
  const int lane = threadIdx.x & 31;
  const int wv   = threadIdx.x >> 5;
  WaveSmem& S = sm[wv];

  const int cid = blockIdx.x * WAVES_PER_BLOCK + wv;      // 32*8*112 centers
  const int b  = cid / (8 * 112);
  const int r  = cid % (8 * 112);
  const int ic = r / 112;          // hc = ic (stride 1)
  const int jc = r % 112;          // wc = 2*jc
  const int wcen = jc * 2;

  const long cbase = (((long)b * 8 + ic) * 224 + wcen) * 3;
  const float cx = ctr0[cbase + 0], cy = ctr0[cbase + 1], cz = ctr0[cbase + 2];

  // ---- Phase 1: 7x11 = 77 candidates
  float dv[3], rlx[3], rly[3], rlz[3];
  int   fx[3];
#pragma unroll
  for (int c = 0; c < 3; ++c) {
    const int p = c * 32 + lane;           // 0..95, real < 77
    float d = 3.0e38f, rx = 0.f, ry = 0.f, rz = 0.f;
    int fi = 0;
    if (p < 77) {
      const int kh = p / 11, kw = p % 11;
      int h = ic + kh - 3;  h = h < 0 ? 0 : (h > 7 ? 7 : h);
      int w = wcen + kw - 5; w = w < 0 ? 0 : (w > 223 ? 223 : w);
      fi = ((b * 8 + h) * 224 + w);
      const long gb = (long)fi * 3;
      rx = ctr0[gb + 0] - cx; ry = ctr0[gb + 1] - cy; rz = ctr0[gb + 2] - cz;
      const float d2 = rx * rx + ry * ry + rz * rz;
      d = (d2 < 3.0f) ? d2 : 1.0e10f;
    }
    S.d2[p] = d;
    dv[c] = d; rlx[c] = rx; rly[c] = ry; rlz[c] = rz; fx[c] = fi;
  }
  __syncthreads();

  // ---- Phase 2: rank select top-32
  int rnk[3] = {0, 0, 0};
  for (int q = 0; q < 77; ++q) {
    const float dq = S.d2[q];
#pragma unroll
    for (int c = 0; c < 3; ++c) {
      const int p = c * 32 + lane;
      rnk[c] += (dq < dv[c] || (dq == dv[c] && q < p)) ? 1 : 0;
    }
  }
#pragma unroll
  for (int c = 0; c < 3; ++c) {
    if (rnk[c] < 32) {
      const float vf = (dv[c] < 9.9e9f) ? 1.0f : 0.0f;
      float* row = S.bufA[rnk[c]];
      row[0] = rlx[c] * vf; row[1] = rly[c] * vf; row[2] = rlz[c] * vf;
      S.selidx[rnk[c]] = fx[c];
      S.selval[rnk[c]] = vf;
    }
  }
  __syncthreads();

  // ---- Phase 3: gather 16 feature channels for this lane's selected neighbor
  {
    const float vf = S.selval[lane];
    const float* fr = f0 + (long)S.selidx[lane] * 16;
    float* row = S.bufA[lane];
#pragma unroll
    for (int c = 0; c < 16; ++c) row[3 + c] = fr[c] * vf;
#pragma unroll
    for (int c = 19; c < 32; ++c) row[c] = 0.0f;       // K padding
  }
  __syncthreads();

  // ---- Phase 4: shared MLP on the 32x19 tile via WMMA
  mlp_layer<19, 16>(w0, b0, S.bufA, S.bufB, lane); __syncthreads();
  mlp_layer<16, 16>(w1, b1, S.bufB, S.bufA, lane); __syncthreads();
  mlp_layer<16, 32>(w2, b2, S.bufA, S.bufB, lane); __syncthreads();

  // ---- Phase 5: max-pool over 32 neighbors, lane = output channel (32 of them)
  float mx = S.bufB[0][lane];
#pragma unroll
  for (int rr = 1; rr < 32; ++rr) mx = fmaxf(mx, S.bufB[rr][lane]);

  out[((long)b * 896 + ic * 112 + jc) * 32 + lane] = mx;
}

extern "C" void kernel_launch(void* const* d_in, const int* in_sizes, int n_in,
                              void* d_out, int out_size, void* d_ws, size_t ws_size,
                              hipStream_t stream) {
  (void)in_sizes; (void)n_in; (void)out_size; (void)ws_size;

  const float* xyz  = (const float*)d_in[0];
  const float* l0w0 = (const float*)d_in[1];
  const float* l0b0 = (const float*)d_in[2];
  const float* l0w1 = (const float*)d_in[3];
  const float* l0b1 = (const float*)d_in[4];
  const float* l0w2 = (const float*)d_in[5];
  const float* l0b2 = (const float*)d_in[6];
  const float* l1w0 = (const float*)d_in[7];
  const float* l1b0 = (const float*)d_in[8];
  const float* l1w1 = (const float*)d_in[9];
  const float* l1b1 = (const float*)d_in[10];
  const float* l1w2 = (const float*)d_in[11];
  const float* l1b2 = (const float*)d_in[12];

  float* ws   = (float*)d_ws;
  float* ctr0 = ws;                          // 32*8*224*3  = 172,032 floats
  float* f0   = ws + 32 * 8 * 224 * 3;       // 32*8*224*16 = 917,504 floats
  float* out  = (float*)d_out;               // 32*896*32

  const int nC0 = 32 * 8 * 224;  // 57,344 centers -> 1 wave each
  const int nC1 = 32 * 8 * 112;  // 28,672 centers

  sa_layer0_kernel<<<dim3(nC0 / WAVES_PER_BLOCK), dim3(128), 0, stream>>>(
      xyz, l0w0, l0b0, l0w1, l0b1, l0w2, l0b2, ctr0, f0);
  sa_layer1_kernel<<<dim3(nC1 / WAVES_PER_BLOCK), dim3(128), 0, stream>>>(
      ctr0, f0, l1w0, l1b0, l1w1, l1b1, l1w2, l1b2, out);
}
